// SimpleGraphConv_80178449482535
// MI455X (gfx1250) — compile-verified, matrix operand order checked by reading, not run on hardware
//
#include <hip/hip_runtime.h>
#include <hip/hip_bf16.h>

typedef float    v8f  __attribute__((ext_vector_type(8)));
typedef float    v2f  __attribute__((ext_vector_type(2)));
typedef _Float16 v16h __attribute__((ext_vector_type(16)));

#define GC_N 64
#define GC_F 7
#define GC_K (GC_N * GC_F)            /* 448 */
#define BATCH_PER_WAVE 16
#define WAVES_PER_BLOCK 8
#define BATCH_PER_BLOCK (BATCH_PER_WAVE * WAVES_PER_BLOCK)  /* 128 */
#define THREADS 256

// Kernel 1: per 16-batch tile per wave, compute y_b = sum_k x[b,k] * v[k]
// (v[k] = W[k % 7]) via WMMA row-sum with all-ones B. Then apply the
// closed-form spectral coefficient f(nb-1) = h0 + h1*lam + h2*lam^2,
// bias, tensor_scale, square, and block-reduce to one partial per block.
__global__ __launch_bounds__(THREADS) void gc_batch_kernel(
    const float* __restrict__ x, const int* __restrict__ nnodes,
    const float* __restrict__ h, const float* __restrict__ W,
    const float* __restrict__ bias, const float* __restrict__ tscale,
    float* __restrict__ partials)
{
    __shared__ float lds_v[GC_K];
    __shared__ float lds_y[BATCH_PER_BLOCK];
    __shared__ float lds_red[THREADS];

    const int tid  = threadIdx.x;
    const int lane = tid & 31;
    const int wave = tid >> 5;          // 0..7
    const int m    = lane & 15;         // M row within 16x16 tile
    const int kh   = lane >> 4;         // 0: K pair {0,1}, 1: K pair {2,3}
    const long long blockStart = (long long)blockIdx.x * BATCH_PER_BLOCK;

    // Expand weight vector: v[k] = W[k % 7]
    for (int k = tid; k < GC_K; k += THREADS) lds_v[k] = W[k % GC_F];
    __syncthreads();

    const float* row =
        x + (blockStart + (long long)(wave * BATCH_PER_WAVE + m)) * (long long)GC_K;
    __builtin_prefetch(row + 224, 0, 1);   // global_prefetch_b8: second half of row

    v8f acc = {};

#if __has_builtin(__builtin_amdgcn_wmma_f32_16x16x4_f32)
    // f32 path: A = 16x4 tile of (x * v), B = 4x16 all-ones -> D rows = y_b.
    v2f bones;
    bones.x = 1.0f;
    bones.y = 1.0f;
    #pragma unroll 8
    for (int kk = 0; kk < GC_K; kk += 4) {
        const int k0 = kk + 2 * kh;    // documented A layout: lane holds K = k0, k0+1
        v2f a;
        a.x = row[k0]     * lds_v[k0];
        a.y = row[k0 + 1] * lds_v[k0 + 1];
        acc = __builtin_amdgcn_wmma_f32_16x16x4_f32(
            false, a, false, bones, (short)0, acc, false, false);
    }
#else
    // Fallback: f16 inputs, f32 accumulate (output saturates; precision ample).
    v16h bones16;
    #pragma unroll
    for (int e = 0; e < 16; ++e) bones16[e] = (_Float16)1.0f;
    for (int kk = 0; kk < GC_K; kk += 32) {
        v16h a16;
        #pragma unroll
        for (int e = 0; e < 16; ++e) {
            // 16-bit A 16x32 layout: K = (e<8?0:16) + kh*8 + (e&7)
            const int K = kk + ((e < 8) ? 0 : 16) + kh * 8 + (e & 7);
            a16[e] = (_Float16)(row[K] * lds_v[K]);
        }
        acc = __builtin_amdgcn_wmma_f32_16x16x32_f16(
            false, a16, false, bones16, (short)0, acc, false, false);
    }
#endif

    // D columns are identical (row sums); lane with N==0 in each half writes
    // its 8 M-rows: lane 0 -> M 0..7, lane 16 -> M 8..15.
    if (m == 0) {
        #pragma unroll
        for (int r = 0; r < 8; ++r)
            lds_y[wave * 16 + kh * 8 + r] = acc[r];
    }
    __syncthreads();

    float sq = 0.0f;
    if (tid < BATCH_PER_BLOCK) {
        const long long b = blockStart + tid;
        const float ysum  = lds_y[tid];                 // = S_b . w
        const float lam   = (float)(nnodes[b] - 1);     // top eigenvalue of J-I
        const float coef  = h[0] + lam * (h[1] + lam * h[2]);
        const float val   = (coef * ysum + bias[0]) * tscale[0];
        sq = val * val;
    }
    lds_red[tid] = sq;
    __syncthreads();
    #pragma unroll
    for (int s = THREADS / 2; s > 0; s >>= 1) {
        if (tid < s) lds_red[tid] += lds_red[tid + s];
        __syncthreads();
    }
    if (tid == 0) partials[blockIdx.x] = lds_red[0];
}

// Kernel 2: one wave32 reduces the per-block partials and applies sigmoid.
__global__ __launch_bounds__(32) void gc_final_kernel(
    const float* __restrict__ partials, int nPartials, float* __restrict__ out)
{
    const int lane = threadIdx.x;
    float v = 0.0f;
    for (int i = lane; i < nPartials; i += 32) v += partials[i];
    #pragma unroll
    for (int off = 16; off > 0; off >>= 1) v += __shfl_down(v, off);
    if (lane == 0) out[0] = 1.0f / (1.0f + __expf(-v));
}

extern "C" void kernel_launch(void* const* d_in, const int* in_sizes, int n_in,
                              void* d_out, int out_size, void* d_ws, size_t ws_size,
                              hipStream_t stream) {
    const float* x  = (const float*)d_in[0];   // [B, 64, 7] f32
    const int*   nn = (const int*)d_in[1];     // [B] i32
    const float* h  = (const float*)d_in[2];   // [3]
    const float* W  = (const float*)d_in[3];   // [1,7]
    const float* b  = (const float*)d_in[4];   // [1]
    const float* ts = (const float*)d_in[5];   // [1]
    float* out      = (float*)d_out;           // scalar f32

    const int B       = in_sizes[0] / GC_K;            // 4096
    const int nBlocks = B / BATCH_PER_BLOCK;           // 32
    float* partials   = (float*)d_ws;                  // 32 floats scratch

    gc_batch_kernel<<<nBlocks, THREADS, 0, stream>>>(x, nn, h, W, b, ts, partials);
    gc_final_kernel<<<1, 32, 0, stream>>>(partials, nBlocks, out);
}
